// MultiHeadAttention_69106023792666
// MI455X (gfx1250) — compile-verified, matrix operand order checked by reading, not run on hardware
//
#include <hip/hip_runtime.h>
#include <hip/hip_bf16.h>

#define B_ 2
#define S_ 2048
#define D_ 1024
#define H_ 16
#define HD_ 64

typedef __bf16 v16bf __attribute__((ext_vector_type(16)));
typedef float v8f __attribute__((ext_vector_type(8)));
typedef unsigned int v4u __attribute__((ext_vector_type(4)));

union FragU {
  struct { v4u lo, hi; } s;
  v16bf b;
};

__device__ __forceinline__ unsigned short f2bf(float x) {
  unsigned int u = __builtin_bit_cast(unsigned int, x);
  u += 0x7fffu + ((u >> 16) & 1u);   // round-to-nearest-even
  return (unsigned short)(u >> 16);
}

// Load a 16x32 bf16 fragment from a row-major [rows x ld] matrix at (row0,k0).
// CDNA5 16-bit A/B layout: lane r=L%16 holds row (row0+r); half=L/16 selects
// the K subrange. Per lane this is exactly two 16B-aligned b128 loads.
__device__ __forceinline__ v16bf load_frag(const unsigned short* p, int ld,
                                           int row0, int k0, int lane) {
  const int r = lane & 15, hh = lane >> 4;
  const unsigned short* rowp = p + (size_t)(row0 + r) * ld + k0 + 8 * hh;
  const v4u* q = (const v4u*)rowp;   // 16B aligned by construction
  FragU f;
  f.s.lo = q[0];                     // dwords 4*hh .. 4*hh+3
  f.s.hi = q[2];                     // dwords 8+4*hh .. 11+4*hh
  return f.b;
}

__device__ __forceinline__ v8f wmma_bf16(v16bf a, v16bf b, v8f c) {
  return __builtin_amdgcn_wmma_f32_16x16x32_bf16(false, a, false, b,
                                                 (short)0, c, false, false);
}

__global__ void cvt_f32_to_bf16(const float* __restrict__ in,
                                unsigned short* __restrict__ out, int n) {
  int i = blockIdx.x * blockDim.x + threadIdx.x;
  if (i < n) out[i] = f2bf(in[i]);
}

// y = x @ W^T  (x: [B*S, D] bf16, W: [D, D] bf16 row-major [n,k]).
// Each wave computes a 32x64 tile: 2 A fragments x 4 B fragments = 8 WMMAs
// per K-step on 12 b128 loads (~21 FLOP/L2-byte).
// grid.z selects Wq/Wk/Wv. Q,K written [b,h,s,hd]; V written transposed [b,h,hd,s].
__global__ void qkv_gemm(const unsigned short* __restrict__ xb,
                         const unsigned short* __restrict__ Wq,
                         const unsigned short* __restrict__ Wk,
                         const unsigned short* __restrict__ Wv,
                         unsigned short* __restrict__ Qb,
                         unsigned short* __restrict__ Kb,
                         unsigned short* __restrict__ VTb) {
  const int lane = threadIdx.x;
  const int m0 = (blockIdx.x * 4 + threadIdx.y) * 32;
  const int n0 = blockIdx.y * 64;
  const int sel = blockIdx.z;
  const unsigned short* W = (sel == 0) ? Wq : (sel == 1) ? Wk : Wv;
  unsigned short* out = (sel == 0) ? Qb : (sel == 1) ? Kb : VTb;

  v8f acc[2][4] = {};
  for (int kk = 0; kk < D_; kk += 32) {
    if (kk + 32 < D_) {
      __builtin_prefetch(xb + (size_t)m0 * D_ + kk + 32, 0, 1);
      __builtin_prefetch(W + (size_t)n0 * D_ + kk + 32, 0, 1);
    }
    const v16bf a0 = load_frag(xb, D_, m0, kk, lane);
    const v16bf a1 = load_frag(xb, D_, m0 + 16, kk, lane);
#pragma unroll
    for (int t = 0; t < 4; ++t) {
      const v16bf b = load_frag(W, D_, n0 + 16 * t, kk, lane);
      acc[0][t] = wmma_bf16(a0, b, acc[0][t]);
      acc[1][t] = wmma_bf16(a1, b, acc[1][t]);
    }
  }

  const int hh = lane >> 4, col = lane & 15;
#pragma unroll
  for (int i = 0; i < 2; ++i) {
#pragma unroll
    for (int t = 0; t < 4; ++t) {
#pragma unroll
      for (int j = 0; j < 8; ++j) {
        const int m = m0 + 16 * i + j + 8 * hh;   // row in [B*S)
        const int n = n0 + 16 * t + col;          // col in [D)
        const int bb = m >> 11, s = m & (S_ - 1);
        const int h = n >> 6, hd = n & (HD_ - 1);
        const size_t idx = (sel == 2)
            ? ((size_t)((bb * H_ + h) * HD_ + hd) * S_ + s)   // V transposed
            : ((size_t)((bb * H_ + h) * S_ + s) * HD_ + hd);  // Q, K
        out[idx] = f2bf(acc[i][t][j]);
      }
    }
  }
}

// Flash-style causal attention. One wave per 16-query tile; key blocks of 32.
// ctx written bf16 into [b, s, h*64+hd] layout for the output projection.
__global__ void attn_kernel(const unsigned short* __restrict__ Qb,
                            const unsigned short* __restrict__ Kb,
                            const unsigned short* __restrict__ VTb,
                            unsigned short* __restrict__ ctxb) {
  const int lane = threadIdx.x;
  const int qt = blockIdx.x * 4 + threadIdx.y;
  const int h = blockIdx.y, bz = blockIdx.z;
  const int q0 = qt * 16;
  const int hh = lane >> 4, col = lane & 15;

  const unsigned short* Q = Qb + (size_t)(bz * H_ + h) * S_ * HD_;
  const unsigned short* K = Kb + (size_t)(bz * H_ + h) * S_ * HD_;
  const unsigned short* VT = VTb + (size_t)(bz * H_ + h) * HD_ * S_;

  __shared__ unsigned int lds_p[4][16][16];     // per-wave 16x32 bf16 P tile
  unsigned int(*tile)[16] = lds_p[threadIdx.y];

  v8f a0 = {}, a1 = {}, a2 = {}, a3 = {};
  float Mrow[8], Lrow[8];
#pragma unroll
  for (int j = 0; j < 8; ++j) { Mrow[j] = -3.0e38f; Lrow[j] = 0.0f; }

  const v16bf qf0 = load_frag(Q, HD_, q0, 0, lane);    // Q tile, d=0..31
  const v16bf qf1 = load_frag(Q, HD_, q0, 32, lane);   // Q tile, d=32..63

  const int kbmax = (q0 + 15) >> 5;
  for (int kb = 0; kb <= kbmax; ++kb) {
    const int k0 = kb * 32;
    // scores: two 16x16 subtiles over 32 keys, K-dim = HD = 64 (2 WMMAs each)
    v8f sa = {}, sb = {};
    sa = wmma_bf16(qf0, load_frag(K, HD_, k0, 0, lane), sa);
    sa = wmma_bf16(qf1, load_frag(K, HD_, k0, 32, lane), sa);
    sb = wmma_bf16(qf0, load_frag(K, HD_, k0 + 16, 0, lane), sb);
    sb = wmma_bf16(qf1, load_frag(K, HD_, k0 + 16, 32, lane), sb);

#pragma unroll
    for (int j = 0; j < 8; ++j) {
      const int qrow = q0 + j + 8 * hh;
      float xa = sa[j] * 0.125f;                 // 1/sqrt(64)
      float xb2 = sb[j] * 0.125f;
      if (k0 + col > qrow) xa = -3.0e38f;        // causal mask
      if (k0 + 16 + col > qrow) xb2 = -3.0e38f;
      float mx = fmaxf(xa, xb2);
#pragma unroll
      for (int off = 8; off; off >>= 1) mx = fmaxf(mx, __shfl_xor(mx, off, 32));
      const float Mn = fmaxf(Mrow[j], mx);
      const float alpha = __expf(Mrow[j] - Mn);
      Mrow[j] = Mn;
      const float pa = __expf(xa - Mn), pb = __expf(xb2 - Mn);
      float rs = pa + pb;
#pragma unroll
      for (int off = 8; off; off >>= 1) rs += __shfl_xor(rs, off, 32);
      Lrow[j] = Lrow[j] * alpha + rs;
      a0[j] *= alpha; a1[j] *= alpha; a2[j] *= alpha; a3[j] *= alpha;
      unsigned short* trow = (unsigned short*)&tile[j + 8 * hh][0];
      trow[col] = f2bf(pa);
      trow[col + 16] = f2bf(pb);
    }

    asm volatile("s_wait_dscnt 0" ::: "memory");  // P stores -> fragment loads

    // P fragment from LDS: two ds_load_b128 per lane
    FragU pf;
    const v4u* q = (const v4u*)&tile[col][hh * 4];
    pf.s.lo = q[0];
    pf.s.hi = q[2];
    const v16bf pfrag = pf.b;

    // ctx += P(16x32) x V(32x64): B-fragment from transposed V, row-major.
    a0 = wmma_bf16(pfrag, load_frag(VT, S_, 0, k0, lane), a0);
    a1 = wmma_bf16(pfrag, load_frag(VT, S_, 16, k0, lane), a1);
    a2 = wmma_bf16(pfrag, load_frag(VT, S_, 32, k0, lane), a2);
    a3 = wmma_bf16(pfrag, load_frag(VT, S_, 48, k0, lane), a3);
  }

#pragma unroll
  for (int j = 0; j < 8; ++j) {
    const float inv = 1.0f / Lrow[j];
    const int s = q0 + j + 8 * hh;
    const size_t rowoff = ((size_t)bz * S_ + s) * D_ + h * HD_;
    ctxb[rowoff + col] = f2bf(a0[j] * inv);
    ctxb[rowoff + 16 + col] = f2bf(a1[j] * inv);
    ctxb[rowoff + 32 + col] = f2bf(a2[j] * inv);
    ctxb[rowoff + 48 + col] = f2bf(a3[j] * inv);
  }
}

// out = ctx @ Wo^T + bo, fp32 result. 32x64 tile per wave.
__global__ void out_gemm(const unsigned short* __restrict__ ctxb,
                         const unsigned short* __restrict__ Wo,
                         const float* __restrict__ bo,
                         float* __restrict__ out) {
  const int lane = threadIdx.x;
  const int m0 = (blockIdx.x * 4 + threadIdx.y) * 32;
  const int n0 = blockIdx.y * 64;

  v8f acc[2][4] = {};
  for (int kk = 0; kk < D_; kk += 32) {
    if (kk + 32 < D_) {
      __builtin_prefetch(ctxb + (size_t)m0 * D_ + kk + 32, 0, 1);
      __builtin_prefetch(Wo + (size_t)n0 * D_ + kk + 32, 0, 1);
    }
    const v16bf a0 = load_frag(ctxb, D_, m0, kk, lane);
    const v16bf a1 = load_frag(ctxb, D_, m0 + 16, kk, lane);
#pragma unroll
    for (int t = 0; t < 4; ++t) {
      const v16bf b = load_frag(Wo, D_, n0 + 16 * t, kk, lane);
      acc[0][t] = wmma_bf16(a0, b, acc[0][t]);
      acc[1][t] = wmma_bf16(a1, b, acc[1][t]);
    }
  }

  const int hh = lane >> 4, col = lane & 15;
#pragma unroll
  for (int i = 0; i < 2; ++i) {
#pragma unroll
    for (int t = 0; t < 4; ++t) {
      const int n = n0 + 16 * t + col;
      const float bias = bo[n];
#pragma unroll
      for (int j = 0; j < 8; ++j) {
        const int m = m0 + 16 * i + j + 8 * hh;
        out[(size_t)m * D_ + n] = acc[i][t][j] + bias;
      }
    }
  }
}

extern "C" void kernel_launch(void* const* d_in, const int* in_sizes, int n_in,
                              void* d_out, int out_size, void* d_ws, size_t ws_size,
                              hipStream_t stream) {
  const float* x = (const float*)d_in[0];
  const float* Wq = (const float*)d_in[1];
  const float* Wk = (const float*)d_in[2];
  const float* Wv = (const float*)d_in[3];
  const float* Wo = (const float*)d_in[4];
  const float* bo = (const float*)d_in[5];
  float* out = (float*)d_out;

  const int NX = B_ * S_ * D_;   // 4,194,304
  const int NW = D_ * D_;        // 1,048,576

  unsigned short* ws = (unsigned short*)d_ws;
  unsigned short* xb = ws;             // [B*S, D] bf16 (reused as ctx later)
  unsigned short* Wqb = xb + NX;
  unsigned short* Wkb = Wqb + NW;
  unsigned short* Wvb = Wkb + NW;
  unsigned short* Wob = Wvb + NW;
  unsigned short* Qb = Wob + NW;       // [b,h,s,hd]
  unsigned short* Kb = Qb + NX;        // [b,h,s,hd]
  unsigned short* VTb = Kb + NX;       // [b,h,hd,s]
  unsigned short* ctxb = xb;           // x dead after qkv_gemm -> reuse

  cvt_f32_to_bf16<<<(NX + 255) / 256, 256, 0, stream>>>(x, xb, NX);
  cvt_f32_to_bf16<<<(NW + 255) / 256, 256, 0, stream>>>(Wq, Wqb, NW);
  cvt_f32_to_bf16<<<(NW + 255) / 256, 256, 0, stream>>>(Wk, Wkb, NW);
  cvt_f32_to_bf16<<<(NW + 255) / 256, 256, 0, stream>>>(Wv, Wvb, NW);
  cvt_f32_to_bf16<<<(NW + 255) / 256, 256, 0, stream>>>(Wo, Wob, NW);

  qkv_gemm<<<dim3((B_ * S_) / 128, D_ / 64, 3), dim3(32, 4), 0, stream>>>(
      xb, Wqb, Wkb, Wvb, Qb, Kb, VTb);

  attn_kernel<<<dim3(S_ / 64, H_, B_), dim3(32, 4), 0, stream>>>(Qb, Kb, VTb, ctxb);

  out_gemm<<<dim3((B_ * S_) / 128, D_ / 64), dim3(32, 4), 0, stream>>>(
      ctxb, Wob, bo, out);
}
